// NetPNA_72945724555680
// MI455X (gfx1250) — compile-verified, matrix operand order checked by reading, not run on hardware
//
#include <hip/hip_runtime.h>

// PNA GNN forward for MI455X (gfx1250, wave32, WMMA).
// All matmuls run on v_wmma_f32_16x16x32_f16. Edge-message GEMM is fused with
// the multi-aggregator scatter (float atomics + ordered-uint min/max atomics)
// so the [E,5,75] message tensor is never materialized. Weight-chunk staging
// into LDS uses GLOBAL_LOAD_ASYNC_TO_LDS_B128 (ASYNCcnt path) when the
// toolchain exposes the builtin, otherwise a vector-register copy. Biases are
// folded into a constant-1 K row of the padded f16 weights. Workspace ~75 MB.

#define N_NODES 8192
#define N_EDGES 262144
#define N_GRAPH 64
#define AVG_DEG_LOG 2.8332133440562162f   // log(17)

typedef _Float16 v16h __attribute__((ext_vector_type(16)));
typedef _Float16 v8h  __attribute__((ext_vector_type(8)));
typedef float    v8f  __attribute__((ext_vector_type(8)));

#if defined(__gfx1250__) && __has_builtin(__builtin_amdgcn_global_load_async_to_lds_b128)
#define ASYNC_LDS 1
// Builtin signature (from compiler diagnostic): arg0 = AS1 pointer to
// int __attribute__((vector_size(16))), arg1 = LDS pointer, then 2 imm ints.
typedef int v4i_gnu __attribute__((vector_size(16)));
typedef __attribute__((address_space(1))) v4i_gnu* as1_v4ip;
typedef __attribute__((address_space(3))) v4i_gnu* as3_v4ip;
#endif

__device__ __forceinline__ void async_copy_b128(const uint4* g, uint4* l) {
#if defined(ASYNC_LDS)
  __builtin_amdgcn_global_load_async_to_lds_b128(
      (as1_v4ip)(const void*)g, (as3_v4ip)(void*)l, 0, 0);
#else
  *l = *g;
#endif
}

__device__ __forceinline__ void async_wait() {
#if defined(ASYNC_LDS)
#if __has_builtin(__builtin_amdgcn_s_wait_asynccnt)
  __builtin_amdgcn_s_wait_asynccnt(0);
#else
  asm volatile("s_wait_asynccnt 0x0" ::: "memory");
#endif
#endif
}

__device__ __forceinline__ unsigned f2ord(float f) {
  unsigned u = __float_as_uint(f);
  return (u & 0x80000000u) ? ~u : (u | 0x80000000u);
}
__device__ __forceinline__ float ord2f(unsigned u) {
  u = (u & 0x80000000u) ? (u & 0x7fffffffu) : ~u;
  return __uint_as_float(u);
}
__device__ __forceinline__ v16h cat8(v8h lo, v8h hi) {
  return __builtin_shufflevector(lo, hi, 0,1,2,3,4,5,6,7,8,9,10,11,12,13,14,15);
}

// ---------------------------------------------------------------- utilities
__global__ void memset32_kernel(unsigned* p, unsigned v, int n) {
  int i = blockIdx.x * blockDim.x + threadIdx.x;
  if (i < n) p[i] = v;
}

__global__ void count_deg_kernel(const int* dst, int* deg) {
  int e = blockIdx.x * blockDim.x + threadIdx.x;
  if (e < N_EDGES) atomicAdd(&deg[dst[e]], 1);
}

__global__ void init_h16_kernel(const int* x, const float* node_emb, _Float16* h16) {
  int i = blockIdx.x * blockDim.x + threadIdx.x;
  if (i >= N_NODES * 80) return;
  int n = i / 80, c = i % 80;
  h16[i] = (c < 75) ? (_Float16)node_emb[x[n] * 75 + c] : (_Float16)0.f;
}

// --------------------------------------------------------- weight repacking
// ee16[4][80]: edge_encoder applied to the 4 possible edge_attr values.
__global__ void pack_ee_kernel(const float* edge_emb, const float* ew, const float* eb,
                               _Float16* ee16) {
  int i = blockIdx.x * blockDim.x + threadIdx.x;
  if (i >= 4 * 80) return;
  int a = i / 80, c = i % 80;
  float v = 0.f;
  if (c < 75) {
    v = eb[c];
    for (int j = 0; j < 50; ++j) v += edge_emb[a * 50 + j] * ew[j * 75 + c];
  }
  ee16[i] = (_Float16)v;
}

// Wpre[384][256] f16, N-major: row n = output dim (t*75+f), col k = input dim.
// k==225 row carries pre_b (A supplies constant 1 there).
__global__ void pack_wpre_kernel(const float* pre_w, const float* pre_b, _Float16* W) {
  int i = blockIdx.x * blockDim.x + threadIdx.x;
  if (i >= 384 * 256) return;
  int n = i / 256, k = i % 256;
  float v = 0.f;
  if (n < 375) {
    int t = n / 75, f = n % 75;
    if (k < 225)        v = pre_w[(t * 225 + k) * 75 + f];
    else if (k == 225)  v = pre_b[t * 75 + f];
  }
  W[i] = (_Float16)v;
}

// Wpost[5][16][992] f16, N-major per tower. k==975 row carries post_b.
__global__ void pack_wpost_kernel(const float* post_w, const float* post_b, _Float16* W) {
  int i = blockIdx.x * blockDim.x + threadIdx.x;
  if (i >= 5 * 16 * 992) return;
  int t = i / (16 * 992), r = i % (16 * 992);
  int nf = r / 992, k = r % 992;
  float v = 0.f;
  if (nf < 15) {
    if (k < 975)        v = post_w[(t * 975 + k) * 15 + nf];
    else if (k == 975)  v = post_b[t * 15 + nf];
  }
  W[i] = (_Float16)v;
}

// Wlin[80][96] f16, N-major. k==80 row carries lin_b.
__global__ void pack_wlin_kernel(const float* lin_w, const float* lin_b, _Float16* W) {
  int i = blockIdx.x * blockDim.x + threadIdx.x;
  if (i >= 80 * 96) return;
  int n = i / 96, k = i % 96;
  float v = 0.f;
  if (n < 75) {
    if (k < 75)        v = lin_w[k * 75 + n];
    else if (k == 80)  v = lin_b[n];
  }
  W[i] = (_Float16)v;
}

// -------------------------------------------- fused edge-message GEMM + scatter
// Block = 256 threads = 8 waves; each wave owns 16 edges (one WMMA M-tile).
// A[16x256] = [h[dst] | h[src] | ee(attr) | 1 | pad] per edge, staged in LDS,
// then held in 64 VGPRs (8 x v16h) for the whole block. Weight chunks of 64
// output dims staged in LDS (async-to-LDS path when available); per N-tile all
// 8 B fragments are preloaded so the 8 WMMAs issue back-to-back. C elements
// are aggregated straight into per-node sum/sq/min/max accumulators.
__global__ __launch_bounds__(256) void edge_msg_kernel(
    const int* __restrict__ srcI, const int* __restrict__ dstI,
    const int* __restrict__ attr, const _Float16* __restrict__ h16,
    const _Float16* __restrict__ ee16, const _Float16* __restrict__ Wpre,
    float* sumA, float* sqA, unsigned* minA, unsigned* maxA) {
  __shared__ _Float16 As[8][16][256];   // 64 KB
  __shared__ _Float16 Bs[64][256];      // 32 KB
  __shared__ int sDst[128];

  int tid = threadIdx.x, w = tid >> 5, lane = tid & 31;
  int eg = blockIdx.x * 128 + w * 16;

  // Stage this wave's 16-edge hcat tile.
  for (int i = lane; i < 16 * 256; i += 32) {
    int el = i >> 8, k = i & 255;
    int e = eg + el;
    _Float16 v = (_Float16)0.f;
    if (k < 75)         v = h16[dstI[e] * 80 + k];
    else if (k < 150)   v = h16[srcI[e] * 80 + (k - 75)];
    else if (k < 225)   v = ee16[attr[e] * 80 + (k - 150)];
    else if (k == 225)  v = (_Float16)1.f;   // bias row
    As[w][el][k] = v;
  }
  if (lane < 16) sDst[w * 16 + lane] = dstI[eg + lane];
  __syncthreads();

  int m = lane & 15;
  int ksel = (lane >> 4) * 8;     // documented 16-bit A-operand lane striping
  int rowoff = (lane >> 4) * 8;

  // Keep the whole A tile (16x256) of this wave in registers.
  v16h af[8];
#pragma unroll
  for (int ks = 0; ks < 8; ++ks) {
    int kb = ks * 32;
    af[ks] = cat8(*(const v8h*)&As[w][m][kb + ksel],
                  *(const v8h*)&As[w][m][kb + 16 + ksel]);
  }

  for (int ch = 0; ch < 6; ++ch) {
    if (ch) __syncthreads();               // all waves done reading prev chunk
    // Stage 64 weight rows (32 KB) cooperatively (async-to-LDS when available).
    const uint4* gsrc = (const uint4*)(Wpre + ch * 64 * 256);
    uint4* ldst = (uint4*)&Bs[0][0];
    for (int i = tid; i < 2048; i += 256) async_copy_b128(gsrc + i, ldst + i);
    async_wait();
    if (ch < 5 && lane == 0)               // prefetch next chunk toward L2
      __builtin_prefetch(Wpre + (ch + 1) * 64 * 256, 0, 1);
    __syncthreads();

    for (int nt = 0; nt < 4; ++nt) {
      // Preload all 8 B fragments, then run the 8-WMMA chain back-to-back.
      v16h bf[8];
#pragma unroll
      for (int ks = 0; ks < 8; ++ks) {
        int kb = ks * 32;
        bf[ks] = cat8(*(const v8h*)&Bs[nt * 16 + m][kb + ksel],
                      *(const v8h*)&Bs[nt * 16 + m][kb + 16 + ksel]);
      }
      v8f acc = {0.f, 0.f, 0.f, 0.f, 0.f, 0.f, 0.f, 0.f};
#pragma unroll
      for (int ks = 0; ks < 8; ++ks)
        acc = __builtin_amdgcn_wmma_f32_16x16x32_f16(
            false, af[ks], false, bf[ks], (short)0, acc, false, false);

      int nbase = ch * 64 + nt * 16 + m;   // output msg dim (t*75+f)
      if (nbase < 375) {
        for (int r = 0; r < 8; ++r) {
          int row = r + rowoff;
          int node = sDst[w * 16 + row];
          float v = acc[r];
          atomicAdd(&sumA[node * 384 + nbase], v);
          atomicAdd(&sqA[node * 384 + nbase], v * v);
          unsigned u = f2ord(v);
          atomicMin(&minA[node * 384 + nbase], u);
          atomicMax(&maxA[node * 384 + nbase], u);
        }
      }
    }
  }
}

// ------------------------------- per-node aggregation finalize -> post_in (f16)
__global__ void finalize_tower_kernel(int t, const _Float16* h16, const int* deg,
                                      const float* sumA, const float* sqA,
                                      const unsigned* minA, const unsigned* maxA,
                                      _Float16* postin) {
  int i = blockIdx.x * blockDim.x + threadIdx.x;
  if (i >= N_NODES * 992) return;
  int n = i / 992, k = i % 992;
  float v;
  if (k < 75) {
    v = (float)h16[n * 80 + k];
  } else if (k < 975) {
    int c = k - 75;
    int g = c / 300, s = (c % 300) / 75, f = c % 75;
    int dim = t * 75 + f;
    int d = deg[n];
    float dc = fmaxf((float)d, 1.f);
    float inv = 1.f / dc;
    float mean = sumA[n * 384 + dim] * inv;
    float stat;
    if (s == 0)      stat = mean;
    else if (s == 1) stat = (d > 0) ? ord2f(minA[n * 384 + dim]) : 0.f;
    else if (s == 2) stat = (d > 0) ? ord2f(maxA[n * 384 + dim]) : 0.f;
    else {
      float msq = sqA[n * 384 + dim] * inv;
      stat = sqrtf(fmaxf(msq - mean * mean, 0.f) + 1e-5f);
    }
    float amp = logf(dc + 1.f) * (1.f / AVG_DEG_LOG);
    float scl = (g == 0) ? 1.f : ((g == 1) ? amp : 1.f / amp);
    v = stat * scl;
  } else {
    v = (k == 975) ? 1.f : 0.f;   // bias row
  }
  postin[(size_t)i] = (_Float16)v;
}

// --------------------------------- per-tower post GEMM: [8192,992] x [992,16]
__global__ __launch_bounds__(256) void gemm_post_kernel(
    const _Float16* __restrict__ postin, const _Float16* __restrict__ Wpost,
    float* towerout, int t) {
  __shared__ _Float16 Bs[16][992];    // 31 KB
  int tid = threadIdx.x, w = tid >> 5, lane = tid & 31;
  const uint4* gsrc = (const uint4*)(Wpost + t * 16 * 992);
  uint4* ldst = (uint4*)&Bs[0][0];
  for (int i = tid; i < 1984; i += 256) async_copy_b128(gsrc + i, ldst + i);
  async_wait();
  __syncthreads();

  int n0 = blockIdx.x * 128 + w * 16;
  int m = lane & 15, ksel = (lane >> 4) * 8, rowoff = (lane >> 4) * 8;
  const _Float16* pa = postin + (size_t)(n0 + m) * 992;
  // Two interleaved accumulator chains to break the 31-deep WMMA dependency.
  v8f acc[2];
  acc[0] = (v8f){0.f,0.f,0.f,0.f,0.f,0.f,0.f,0.f};
  acc[1] = (v8f){0.f,0.f,0.f,0.f,0.f,0.f,0.f,0.f};
#pragma unroll
  for (int ks = 0; ks < 31; ++ks) {
    int kb = ks * 32;
    v16h a = cat8(*(const v8h*)(pa + kb + ksel),
                  *(const v8h*)(pa + kb + 16 + ksel));
    v16h b = cat8(*(const v8h*)&Bs[m][kb + ksel],
                  *(const v8h*)&Bs[m][kb + 16 + ksel]);
    acc[ks & 1] = __builtin_amdgcn_wmma_f32_16x16x32_f16(
        false, a, false, b, (short)0, acc[ks & 1], false, false);
  }
  for (int r = 0; r < 8; ++r) {
    int node = n0 + r + rowoff;
    towerout[node * 80 + t * 16 + m] = acc[0][r] + acc[1][r];
  }
}

// repack tower outputs [N][5][16] -> contiguous 75 + bias-1 at col 80, f16 [N][96]
__global__ void pack_towerout_kernel(const float* towerout, _Float16* towerpk) {
  int i = blockIdx.x * blockDim.x + threadIdx.x;
  if (i >= N_NODES * 96) return;
  int n = i / 96, k = i % 96;
  float v = 0.f;
  if (k < 75)       v = towerout[n * 80 + (k / 15) * 16 + (k % 15)];
  else if (k == 80) v = 1.f;   // lin bias row
  towerpk[i] = (_Float16)v;
}

// ---------------------------- lin GEMM [8192,96]x[96,80] + BN stats via atomics
__global__ __launch_bounds__(256) void gemm_lin_kernel(
    const _Float16* __restrict__ towerpk, const _Float16* __restrict__ Wlin,
    float* outpre, float* bnsum, float* bnsq) {
  __shared__ _Float16 Bs[80][96];     // 15 KB
  int tid = threadIdx.x, w = tid >> 5, lane = tid & 31;
  const uint4* gsrc = (const uint4*)Wlin;
  uint4* ldst = (uint4*)&Bs[0][0];
  for (int i = tid; i < 960; i += 256) async_copy_b128(gsrc + i, ldst + i);
  async_wait();
  __syncthreads();

  int n0 = blockIdx.x * 128 + w * 16;
  int m = lane & 15, ksel = (lane >> 4) * 8, rowoff = (lane >> 4) * 8;
  const _Float16* pa = towerpk + (size_t)(n0 + m) * 96;
  v8f acc[5];
#pragma unroll
  for (int nt = 0; nt < 5; ++nt) acc[nt] = (v8f){0.f,0.f,0.f,0.f,0.f,0.f,0.f,0.f};
#pragma unroll
  for (int ks = 0; ks < 3; ++ks) {
    int kb = ks * 32;
    v16h a = cat8(*(const v8h*)(pa + kb + ksel),
                  *(const v8h*)(pa + kb + 16 + ksel));
#pragma unroll
    for (int nt = 0; nt < 5; ++nt) {
      v16h b = cat8(*(const v8h*)&Bs[nt * 16 + m][kb + ksel],
                    *(const v8h*)&Bs[nt * 16 + m][kb + 16 + ksel]);
      acc[nt] = __builtin_amdgcn_wmma_f32_16x16x32_f16(
          false, a, false, b, (short)0, acc[nt], false, false);
    }
  }
  for (int nt = 0; nt < 5; ++nt) {
    int col = nt * 16 + m;
    for (int r = 0; r < 8; ++r) {
      int node = n0 + r + rowoff;
      float v = acc[nt][r];
      outpre[node * 80 + col] = v;
      if (col < 75) {
        atomicAdd(&bnsum[col], v);
        atomicAdd(&bnsq[col], v * v);
      }
    }
  }
}

__global__ void bn_finalize_kernel(const float* bnsum, const float* bnsq,
                                   const float* gamma, const float* beta,
                                   float* bnscale, float* bnshift) {
  int c = threadIdx.x;
  if (c >= 75) return;
  float mu = bnsum[c] * (1.f / N_NODES);
  float var = bnsq[c] * (1.f / N_NODES) - mu * mu;
  float s = gamma[c] * rsqrtf(var + 1e-5f);
  bnscale[c] = s;
  bnshift[c] = beta[c] - mu * s;
}

__global__ void apply_bn_kernel(const float* outpre, const float* bnscale,
                                const float* bnshift, _Float16* h16) {
  int i = blockIdx.x * blockDim.x + threadIdx.x;
  if (i >= N_NODES * 80) return;
  int c = i % 80;
  if (c < 75) {
    float v = outpre[i] * bnscale[c] + bnshift[c];
    h16[i] = (_Float16)fmaxf(v, 0.f);
  } else {
    h16[i] = (_Float16)0.f;
  }
}

// --------------------------------------------------------- pooling + final MLP
__global__ void pool_kernel(const int* batch, const _Float16* h16, float* pooled) {
  int n = blockIdx.x * blockDim.x + threadIdx.x;
  if (n >= N_NODES) return;
  int g = batch[n];
  for (int c = 0; c < 75; ++c)
    atomicAdd(&pooled[g * 80 + c], (float)h16[n * 80 + c]);
}

__global__ void mlp_kernel(const float* pooled,
                           const float* w1, const float* b1,
                           const float* w2, const float* b2,
                           const float* w3, const float* b3, float* out) {
  int g = threadIdx.x;
  if (g >= N_GRAPH) return;
  float y1[50], y2[25];
  for (int j = 0; j < 50; ++j) {
    float s = b1[j];
    for (int c = 0; c < 75; ++c) s += pooled[g * 80 + c] * w1[c * 50 + j];
    y1[j] = fmaxf(s, 0.f);
  }
  for (int j = 0; j < 25; ++j) {
    float s = b2[j];
    for (int c = 0; c < 50; ++c) s += y1[c] * w2[c * 25 + j];
    y2[j] = fmaxf(s, 0.f);
  }
  float o = b3[0];
  for (int c = 0; c < 25; ++c) o += y2[c] * w3[c];
  out[g] = o;
}

// =============================================================== entry point
extern "C" void kernel_launch(void* const* d_in, const int* in_sizes, int n_in,
                              void* d_out, int out_size, void* d_ws, size_t ws_size,
                              hipStream_t stream) {
  const int*   x        = (const int*)  d_in[0];
  const int*   eidx     = (const int*)  d_in[1];
  const int*   eattr    = (const int*)  d_in[2];
  const int*   batch    = (const int*)  d_in[3];
  const float* node_emb = (const float*)d_in[4];
  const float* edge_emb = (const float*)d_in[5];
  const float* enc_w    = (const float*)d_in[6];
  const float* enc_b    = (const float*)d_in[7];
  const float* pre_w    = (const float*)d_in[8];
  const float* pre_b    = (const float*)d_in[9];
  const float* post_w   = (const float*)d_in[10];
  const float* post_b   = (const float*)d_in[11];
  const float* lin_w    = (const float*)d_in[12];
  const float* lin_b    = (const float*)d_in[13];
  const float* gamma    = (const float*)d_in[14];
  const float* beta     = (const float*)d_in[15];
  const float* w1 = (const float*)d_in[16]; const float* b1 = (const float*)d_in[17];
  const float* w2 = (const float*)d_in[18]; const float* b2 = (const float*)d_in[19];
  const float* w3 = (const float*)d_in[20]; const float* b3 = (const float*)d_in[21];
  const int* srcI = eidx;
  const int* dstI = eidx + N_EDGES;

  char* base = (char*)d_ws;
  size_t off = 0;
  auto carve = [&](size_t bytes) -> void* {
    void* p = base + off;
    off = (off + bytes + 255) & ~(size_t)255;
    return p;
  };
  _Float16* h16     = (_Float16*)carve((size_t)N_NODES * 80 * 2);
  _Float16* ee16    = (_Float16*)carve(4 * 80 * 2);
  _Float16* Wpre    = (_Float16*)carve(384 * 256 * 2);
  _Float16* Wpost   = (_Float16*)carve(5 * 16 * 992 * 2);
  _Float16* Wlin    = (_Float16*)carve(80 * 96 * 2);
  float*    sumA    = (float*)   carve((size_t)N_NODES * 384 * 4);
  float*    sqA     = (float*)   carve((size_t)N_NODES * 384 * 4);
  unsigned* minA    = (unsigned*)carve((size_t)N_NODES * 384 * 4);
  unsigned* maxA    = (unsigned*)carve((size_t)N_NODES * 384 * 4);
  int*      deg     = (int*)     carve((size_t)N_NODES * 4);
  _Float16* postin  = (_Float16*)carve((size_t)N_NODES * 992 * 2);
  float*    towerout= (float*)   carve((size_t)N_NODES * 80 * 4);
  _Float16* towerpk = (_Float16*)carve((size_t)N_NODES * 96 * 2);
  float*    outpre  = (float*)   carve((size_t)N_NODES * 80 * 4);
  float*    bnsum   = (float*)   carve(80 * 4);
  float*    bnsq    = (float*)   carve(80 * 4);
  float*    bnscale = (float*)   carve(80 * 4);
  float*    bnshift = (float*)   carve(80 * 4);
  float*    pooled  = (float*)   carve((size_t)N_GRAPH * 80 * 4);

  auto gz = [](int n) { return (n + 255) / 256; };

  memset32_kernel<<<gz(N_NODES), 256, 0, stream>>>((unsigned*)deg, 0u, N_NODES);
  count_deg_kernel<<<gz(N_EDGES), 256, 0, stream>>>(dstI, deg);
  init_h16_kernel<<<gz(N_NODES * 80), 256, 0, stream>>>(x, node_emb, h16);

  for (int L = 0; L < 4; ++L) {
    pack_ee_kernel  <<<2, 256, 0, stream>>>(edge_emb, enc_w + L * 3750, enc_b + L * 75, ee16);
    pack_wpre_kernel<<<gz(384 * 256), 256, 0, stream>>>(pre_w + L * 84375, pre_b + L * 375, Wpre);
    pack_wpost_kernel<<<gz(5 * 16 * 992), 256, 0, stream>>>(post_w + L * 73125, post_b + L * 75, Wpost);
    pack_wlin_kernel<<<gz(80 * 96), 256, 0, stream>>>(lin_w + L * 5625, lin_b + L * 75, Wlin);

    const int NA = N_NODES * 384;
    memset32_kernel<<<gz(NA), 256, 0, stream>>>((unsigned*)sumA, 0u, NA);
    memset32_kernel<<<gz(NA), 256, 0, stream>>>((unsigned*)sqA, 0u, NA);
    memset32_kernel<<<gz(NA), 256, 0, stream>>>(minA, 0xFFFFFFFFu, NA);
    memset32_kernel<<<gz(NA), 256, 0, stream>>>(maxA, 0u, NA);

    edge_msg_kernel<<<N_EDGES / 128, 256, 0, stream>>>(
        srcI, dstI, eattr, h16, ee16, Wpre, sumA, sqA, minA, maxA);

    for (int t = 0; t < 5; ++t) {
      finalize_tower_kernel<<<gz(N_NODES * 992), 256, 0, stream>>>(
          t, h16, deg, sumA, sqA, minA, maxA, postin);
      gemm_post_kernel<<<N_NODES / 128, 256, 0, stream>>>(postin, Wpost, towerout, t);
    }

    pack_towerout_kernel<<<gz(N_NODES * 96), 256, 0, stream>>>(towerout, towerpk);
    memset32_kernel<<<1, 256, 0, stream>>>((unsigned*)bnsum, 0u, 80);
    memset32_kernel<<<1, 256, 0, stream>>>((unsigned*)bnsq, 0u, 80);
    gemm_lin_kernel<<<N_NODES / 128, 256, 0, stream>>>(towerpk, Wlin, outpre, bnsum, bnsq);
    bn_finalize_kernel<<<1, 128, 0, stream>>>(bnsum, bnsq, gamma + L * 75, beta + L * 75,
                                              bnscale, bnshift);
    apply_bn_kernel<<<gz(N_NODES * 80), 256, 0, stream>>>(outpre, bnscale, bnshift, h16);
  }

  memset32_kernel<<<gz(N_GRAPH * 80), 256, 0, stream>>>((unsigned*)pooled, 0u, N_GRAPH * 80);
  pool_kernel<<<gz(N_NODES), 256, 0, stream>>>(batch, h16, pooled);
  mlp_kernel<<<1, 64, 0, stream>>>(pooled, w1, b1, w2, b2, w3, b3, (float*)d_out);
}
